// PointNet2SegModel_80083960201800
// MI455X (gfx1250) — matrix operand order repaired and checked by reading, not compiled, not run on hardware
//
#include <hip/hip_runtime.h>
#include <hip/hip_bf16.h>

typedef __attribute__((ext_vector_type(16))) _Float16 v16h;
typedef __attribute__((ext_vector_type(8)))  float    v8f;

#define BN_EPS 1e-5f

// ------------------------------------------------------------------ split xyz / feats
__global__ void split_kernel(const float* __restrict__ in, float* __restrict__ xyz,
                             float* __restrict__ fts, int total, int Cin) {
  int i = blockIdx.x * blockDim.x + threadIdx.x;
  if (i >= total) return;
  const float* f = in + (size_t)i * Cin;
  float* x = xyz + (size_t)i * 3;
  x[0] = f[0]; x[1] = f[1]; x[2] = f[2];
  float* o = fts + (size_t)i * (Cin - 3);
  for (int c = 3; c < Cin; ++c) o[c - 3] = f[c];
}

// ------------------------------------------------------------------ gather points by index
__global__ void gather_kernel(const float* __restrict__ src, const int* __restrict__ idx,
                              float* __restrict__ out, int Nsrc, int S, int C, int total) {
  int gid = blockIdx.x * blockDim.x + threadIdx.x;
  if (gid >= total) return;
  int c = gid % C; int s = (gid / C) % S; int b = gid / (C * S);
  out[gid] = src[((size_t)b * Nsrc + idx[b * S + s]) * C + c];
}

// ------------------------------------------------------------------ column copy / broadcast / pad for concat buffers
__global__ void copy_cols_kernel(const float* __restrict__ src, float* __restrict__ dst,
                                 int Cs, int Cout, int coff, int total) {
  int gid = blockIdx.x * blockDim.x + threadIdx.x;
  if (gid >= total) return;
  int c = gid % Cs; int r = gid / Cs;
  dst[(size_t)r * Cout + coff + c] = src[gid];
}

__global__ void bcast_cols_kernel(const float* __restrict__ src, float* __restrict__ dst,
                                  int C2, int Cout, int coff, int rpb, int total) {
  int gid = blockIdx.x * blockDim.x + threadIdx.x;
  if (gid >= total) return;
  int c = gid % C2; int r = gid / C2; int b = r / rpb;
  dst[(size_t)r * Cout + coff + c] = src[(size_t)b * C2 + c];
}

__global__ void zero_pad_kernel(float* __restrict__ dst, int K, int Kp, int total) {
  int gid = blockIdx.x * blockDim.x + threadIdx.x;
  if (gid >= total) return;
  int pc = Kp - K; int r = gid / pc; int c = K + gid % pc;
  dst[(size_t)r * Kp + c] = 0.f;
}

__global__ void mul_kernel(float* __restrict__ x, const float* __restrict__ a, int total) {
  int i = blockIdx.x * blockDim.x + threadIdx.x;
  if (i >= total) return;
  x[i] *= a[i];
}

// ------------------------------------------------------------------ max over neighbor axis
__global__ void maxpool_kernel(const float* __restrict__ X, float* __restrict__ Y,
                               int S, int ns, int C, int total) {
  int gid = blockIdx.x * blockDim.x + threadIdx.x;
  if (gid >= total) return;
  int c = gid % C; int s = (gid / C) % S; int b = gid / (C * S);
  const float* x = X + (((size_t)b * S + s) * ns) * C + c;
  float m = -3.4e38f;
  for (int k = 0; k < ns; ++k) m = fmaxf(m, x[(size_t)k * C]);
  Y[gid] = m;
}

// ------------------------------------------------------------------ weight repack f32(O,K) -> f16(O,Kp), zero-padded
__global__ void pack_w_kernel(const float* __restrict__ W, _Float16* __restrict__ Wh,
                              int K, int Kp, int total) {
  int gid = blockIdx.x * blockDim.x + threadIdx.x;
  if (gid >= total) return;
  int k = gid % Kp; int o = gid / Kp;
  Wh[gid] = (k < K) ? (_Float16)W[(size_t)o * K + k] : (_Float16)0.f;
}

// fold bias+BN into per-channel scale/shift: y = acc*s + t
__global__ void prep_bn_kernel(const float* __restrict__ b, const float* __restrict__ gamma,
                               const float* __restrict__ beta, const float* __restrict__ mean,
                               const float* __restrict__ var, float* __restrict__ sv,
                               float* __restrict__ tv, int O) {
  int o = blockIdx.x * blockDim.x + threadIdx.x;
  if (o >= O) return;
  float s = gamma[o] * rsqrtf(var[o] + BN_EPS);
  sv[o] = s;
  tv[o] = beta[o] + (b[o] - mean[o]) * s;
}

// ------------------------------------------------------------------ farthest point sampling (1 block per batch)
__global__ void fps_kernel(const float* __restrict__ xyz, int N, int npoint, int* __restrict__ out) {
  const int T = 256;
  int b = blockIdx.x;
  int tid = threadIdx.x;
  const float* p = xyz + (size_t)b * N * 3;
  __shared__ float sd[T];
  __shared__ int   si[T];
  __shared__ float cent[3];
  int chunk = N / T;                 // N is a multiple of 256 at every level
  float dist[32];
#pragma unroll
  for (int i = 0; i < 32; ++i) dist[i] = 1e10f;
  int farthest = 0;
  for (int it = 0; it < npoint; ++it) {
    if (tid == 0) {
      out[b * npoint + it] = farthest;
      cent[0] = p[(size_t)farthest * 3 + 0];
      cent[1] = p[(size_t)farthest * 3 + 1];
      cent[2] = p[(size_t)farthest * 3 + 2];
    }
    __syncthreads();
    float cx = cent[0], cy = cent[1], cz = cent[2];
    float best = -1.0f; int bi = 0;
#pragma unroll
    for (int i = 0; i < 32; ++i) {
      if (i < chunk) {
        int n = tid * chunk + i;
        float dx = p[(size_t)n*3+0]-cx, dy = p[(size_t)n*3+1]-cy, dz = p[(size_t)n*3+2]-cz;
        float d = dx*dx + dy*dy + dz*dz;
        if (d < dist[i]) dist[i] = d;
        if (dist[i] > best) { best = dist[i]; bi = n; }   // strict > keeps lowest index on ties
      }
    }
    sd[tid] = best; si[tid] = bi;
    __syncthreads();
    for (int o = T / 2; o > 0; o >>= 1) {
      if (tid < o) {
        if (sd[tid + o] > sd[tid]) { sd[tid] = sd[tid + o]; si[tid] = si[tid + o]; }
      }
      __syncthreads();
    }
    farthest = si[0];
    __syncthreads();
  }
}

// ------------------------------------------------------------------ kNN: 1 block per query, distances in LDS
__global__ void knn_kernel(const float* __restrict__ q, const float* __restrict__ p,
                           int N, int S, int ns, int* __restrict__ out) {
  int bs = blockIdx.x; int b = bs / S;
  int tid = threadIdx.x;
  extern __shared__ float smem[];
  float* d  = smem;                 // N floats
  float* rd = smem + N;             // 256 floats
  int*   ri = (int*)(rd + 256);     // 256 ints
  const float* qq = q + (size_t)bs * 3;
  float qx = qq[0], qy = qq[1], qz = qq[2];
  const float* pp = p + (size_t)b * N * 3;
  for (int n = tid; n < N; n += 256) {
    float dx = pp[(size_t)n*3+0]-qx, dy = pp[(size_t)n*3+1]-qy, dz = pp[(size_t)n*3+2]-qz;
    d[n] = dx*dx + dy*dy + dz*dz;
  }
  __syncthreads();
  for (int j = 0; j < ns; ++j) {
    float best = 3.4e38f; int bi = N;
    for (int n = tid; n < N; n += 256) {
      float v = d[n];
      if (v < best) { best = v; bi = n; }
    }
    rd[tid] = best; ri[tid] = bi;
    __syncthreads();
    for (int o = 128; o > 0; o >>= 1) {
      if (tid < o) {
        float ov = rd[tid + o]; int oi = ri[tid + o];
        if (ov < rd[tid] || (ov == rd[tid] && oi < ri[tid])) { rd[tid] = ov; ri[tid] = oi; }
      }
      __syncthreads();
    }
    if (tid == 0) { out[(size_t)bs * ns + j] = ri[0]; d[ri[0]] = 3.4e38f; }
    __syncthreads();
  }
}

// ------------------------------------------------------------------ grouped rows (dx,dy,dz, feats), zero-padded to Cout
__global__ void group_kernel(const float* __restrict__ xyz, const float* __restrict__ fts,
                             const float* __restrict__ nxyz, const int* __restrict__ knn,
                             float* __restrict__ out, int N, int S, int ns, int C, int Cout,
                             int total) {
  int gid = blockIdx.x * blockDim.x + threadIdx.x;
  if (gid >= total) return;
  int s = (gid / ns) % S; int b = gid / (ns * S);
  int src = knn[gid];
  const float* pc = xyz  + ((size_t)b * N + src) * 3;
  const float* qc = nxyz + ((size_t)b * S + s) * 3;
  float* o = out + (size_t)gid * Cout;
  o[0] = pc[0] - qc[0]; o[1] = pc[1] - qc[1]; o[2] = pc[2] - qc[2];
  const float* f = fts + ((size_t)b * N + src) * C;
  for (int c = 0; c < C; ++c) o[3 + c] = f[c];
  for (int c = 3 + C; c < Cout; ++c) o[c] = 0.f;
}

// ------------------------------------------------------------------ 3-NN inverse-distance interpolation (writes concat cols)
__global__ void interp_kernel(const float* __restrict__ x1, const float* __restrict__ x2,
                              const float* __restrict__ p2, float* __restrict__ out,
                              int N1, int N2, int C2, int Cout, int coff, int total) {
  int gid = blockIdx.x * blockDim.x + threadIdx.x;
  if (gid >= total) return;
  int b = gid / N1;
  const float* q = x1 + (size_t)gid * 3;
  float qx = q[0], qy = q[1], qz = q[2];
  const float* pp = x2 + (size_t)b * N2 * 3;
  float b0 = 3.4e38f, b1 = 3.4e38f, b2 = 3.4e38f; int i0 = 0, i1 = 0, i2 = 0;
  for (int m = 0; m < N2; ++m) {
    float dx = pp[(size_t)m*3+0]-qx, dy = pp[(size_t)m*3+1]-qy, dz = pp[(size_t)m*3+2]-qz;
    float dd = dx*dx + dy*dy + dz*dz;
    if (dd < b0)      { b2=b1; i2=i1; b1=b0; i1=i0; b0=dd; i0=m; }
    else if (dd < b1) { b2=b1; i2=i1; b1=dd; i1=m; }
    else if (dd < b2) { b2=dd; i2=m; }
  }
  float r0 = 1.f/(b0 + 1e-8f), r1 = 1.f/(b1 + 1e-8f), r2 = 1.f/(b2 + 1e-8f);
  float rs = r0 + r1 + r2;
  float w0 = r0/rs, w1 = r1/rs, w2 = r2/rs;
  const float* a0 = p2 + ((size_t)b * N2 + i0) * C2;
  const float* a1 = p2 + ((size_t)b * N2 + i1) * C2;
  const float* a2 = p2 + ((size_t)b * N2 + i2) * C2;
  float* o = out + (size_t)gid * Cout + coff;
  for (int c = 0; c < C2; ++c) o[c] = a0[c]*w0 + a1[c]*w1 + a2[c]*w2;
}

// ------------------------------------------------------------------ small VALU linear (attention / final classifier)
// mode 0: x.W^T + b ; mode 1: + BN + ReLU ; mode 2: sigmoid(x.W^T + b)
__global__ void small_linear_kernel(const float* __restrict__ X, const float* __restrict__ W,
                                    const float* __restrict__ bias, const float* __restrict__ gamma,
                                    const float* __restrict__ beta, const float* __restrict__ mean,
                                    const float* __restrict__ var, float* __restrict__ Y,
                                    int R, int K, int O, int mode) {
  int gid = blockIdx.x * blockDim.x + threadIdx.x;
  if (gid >= R * O) return;
  int o = gid % O; int r = gid / O;
  const float* x = X + (size_t)r * K;
  const float* w = W + (size_t)o * K;
  float acc = bias[o];
  for (int k = 0; k < K; ++k) acc += x[k] * w[k];
  if (mode == 1) {
    float s = gamma[o] * rsqrtf(var[o] + BN_EPS);
    acc = (acc - mean[o]) * s + beta[o];
    acc = fmaxf(acc, 0.f);
  } else if (mode == 2) {
    acc = 1.f / (1.f + __expf(-acc));
  }
  Y[gid] = acc;
}

// ------------------------------------------------------------------ WMMA GEMM, branch-free hot loop
// Invariants (enforced by host): R%16==0, K%32==0 (padded layout), O%64==0.
// One wave computes a 16x64 tile: A fragment shared across 4 WMMAs.
// X: f32 (R,K);  Wh: f16 (O,K) prepacked;  y = relu(acc*s[o] + t[o]).
__global__ void gemm_bn_relu_wmma(const float* __restrict__ X, const _Float16* __restrict__ Wh,
                                  const float* __restrict__ sv, const float* __restrict__ tv,
                                  float* __restrict__ Y, int R, int K, int O, int relu) {
  int tilesN = O >> 6;
  int wave = blockIdx.x * 8 + ((int)threadIdx.x >> 5);
  if (wave >= (R >> 4) * tilesN) return;        // whole-wave exit: EXEC all-ones for WMMA
  int tm = wave / tilesN, tn = wave % tilesN;
  int lane = (int)threadIdx.x & 31;
  int half = lane >> 4;
  int idx  = lane & 15;
  // A: row (tm*16+idx); per k-tile lane covers K = [16h..16h+16) contiguous
  const float* xrow = X + (size_t)(tm * 16 + idx) * K;
  // B: column (tn*64 + n*16 + idx); per k-tile lane covers K = [16h..16h+16) contiguous f16
  const _Float16* wbase = Wh + (size_t)(tn * 64 + idx) * K + (half << 4);
  v8f acc[4] = {{}, {}, {}, {}};
  for (int kt = 0; kt < K; kt += 32) {
    const float* xa = xrow + kt + (half << 4);
    float4 x0 = *(const float4*)(xa + 0);
    float4 x1 = *(const float4*)(xa + 4);
    float4 x2 = *(const float4*)(xa + 8);
    float4 x3 = *(const float4*)(xa + 12);
    __builtin_prefetch(xa + 32, 0, 1);          // global_prefetch_b8: next K-tile of X
    v16h a;
    a[0]=(_Float16)x0.x;  a[1]=(_Float16)x0.y;  a[2]=(_Float16)x0.z;  a[3]=(_Float16)x0.w;
    a[4]=(_Float16)x1.x;  a[5]=(_Float16)x1.y;  a[6]=(_Float16)x1.z;  a[7]=(_Float16)x1.w;
    a[8]=(_Float16)x2.x;  a[9]=(_Float16)x2.y;  a[10]=(_Float16)x2.z; a[11]=(_Float16)x2.w;
    a[12]=(_Float16)x3.x; a[13]=(_Float16)x3.y; a[14]=(_Float16)x3.z; a[15]=(_Float16)x3.w;
#pragma unroll
    for (int n = 0; n < 4; ++n) {
      v16h b = *(const v16h*)(wbase + (size_t)(n << 4) * K + kt);   // 32B aligned
      acc[n] = __builtin_amdgcn_wmma_f32_16x16x32_f16(false, a, false, b,
                                                      (short)0, acc[n], false, false);
    }
  }
#pragma unroll
  for (int n = 0; n < 4; ++n) {
    int bcol = tn * 64 + (n << 4) + idx;
    float s = sv[bcol], t = tv[bcol];
#pragma unroll
    for (int r = 0; r < 8; ++r) {
      int orow = tm * 16 + (half << 3) + r;     // C/D layout: VGPR r -> row r+8*half, col=idx
      float v = acc[n][r] * s + t;
      if (relu) v = fmaxf(v, 0.f);
      Y[(size_t)orow * O + bcol] = v;
    }
  }
}

// ------------------------------------------------------------------ host side
struct Layer { const float *W, *b, *beta, *gamma, *mean, *var; };
static inline Layer L(void* const* d_in, int base) {
  Layer l;
  l.W     = (const float*)d_in[base + 0];
  l.b     = (const float*)d_in[base + 1];
  l.beta  = (const float*)d_in[base + 2];
  l.gamma = (const float*)d_in[base + 3];
  l.mean  = (const float*)d_in[base + 4];
  l.var   = (const float*)d_in[base + 5];
  return l;
}

struct GL { const _Float16* Wh; const float* s; const float* t; };

#define EW(total) <<<((total) + 255) / 256, 256, 0, stream>>>

extern "C" void kernel_launch(void* const* d_in, const int* in_sizes, int n_in,
                              void* d_out, int out_size, void* d_ws, size_t ws_size,
                              hipStream_t stream) {
  (void)in_sizes; (void)n_in; (void)out_size; (void)ws_size;
  const int Bn = 4, N = 8192, INCH = 8, CIN = 11;

  const float* features = (const float*)d_in[0];
  // JAX pytree (sorted-key) flattening order; layer dicts flatten W,b,beta,gamma,mean,var.
  Layer attn_l1 = L(d_in, 1);
  const float* attn2_W = (const float*)d_in[7];
  const float* attn2_b = (const float*)d_in[8];
  Layer fp1_0 = L(d_in, 9),   fp1_1 = L(d_in, 15);
  Layer fp2_0 = L(d_in, 21),  fp2_1 = L(d_in, 27);
  Layer fp3_0 = L(d_in, 33),  fp3_1 = L(d_in, 39);
  Layer fp4_0 = L(d_in, 45),  fp4_1 = L(d_in, 51);
  Layer head_l1 = L(d_in, 57);
  const float* head2_W = (const float*)d_in[63];
  const float* head2_b = (const float*)d_in[64];
  Layer sa1_0 = L(d_in, 65),  sa1_1 = L(d_in, 71),  sa1_2 = L(d_in, 77);
  Layer sa2_0 = L(d_in, 83),  sa2_1 = L(d_in, 89),  sa2_2 = L(d_in, 95);
  Layer sa3_0 = L(d_in, 101), sa3_1 = L(d_in, 107), sa3_2 = L(d_in, 113);
  Layer sa4_0 = L(d_in, 119), sa4_1 = L(d_in, 125), sa4_2 = L(d_in, 131);

  // ---- workspace bump allocator (bytes, 256B aligned blocks) ----
  char* wsb = (char*)d_ws;
  size_t off = 0;
  auto allocb = [&](size_t bytes) -> void* {
    off = (off + 255) & ~(size_t)255;
    void* p = wsb + off;
    off += bytes;
    return p;
  };
  auto allocf = [&](size_t n) -> float* { return (float*)allocb(n * sizeof(float)); };

  float* xyz    = allocf((size_t)Bn * N * 3);
  float* feats  = allocf((size_t)Bn * N * INCH);
  float* l1_xyz = allocf((size_t)Bn * 2048 * 3);
  float* l2_xyz = allocf((size_t)Bn * 512 * 3);
  float* l3_xyz = allocf((size_t)Bn * 128 * 3);
  float* l1_pts = allocf((size_t)Bn * 2048 * 128);
  float* l2_pts = allocf((size_t)Bn * 512 * 256);
  float* l3_pts = allocf((size_t)Bn * 128 * 512);
  float* l4_pts = allocf((size_t)Bn * 1024);
  float* attn_h = allocf((size_t)Bn * 512);
  float* attn_a = allocf((size_t)Bn * 1024);
  float* l3f    = allocf((size_t)Bn * 128 * 512);
  float* l2f    = allocf((size_t)Bn * 512 * 256);
  float* l1f    = allocf((size_t)Bn * 2048 * 128);
  int*   fpsidx = (int*)allocb((size_t)Bn * 2048 * 4);
  int*   knnidx = (int*)allocb((size_t)Bn * 2048 * 32 * 4);
  float* bufA   = allocf((size_t)33554432);
  float* bufB   = allocf((size_t)33554432);

  // ---- prepack all WMMA-GEMM weights to f16 (stride Kp) + fold BN ----
  auto prep = [&](const Layer& l, int K, int Kp, int O) -> GL {
    _Float16* wh = (_Float16*)allocb((size_t)O * Kp * sizeof(_Float16));
    float* sv = allocf(O);
    float* tv = allocf(O);
    int tot = O * Kp;
    pack_w_kernel EW(tot) (l.W, wh, K, Kp, tot);
    prep_bn_kernel EW(O) (l.b, l.gamma, l.beta, l.mean, l.var, sv, tv, O);
    GL g; g.Wh = wh; g.s = sv; g.t = tv; return g;
  };
  GL g_sa1_0 = prep(sa1_0, 11, 32, 64);
  GL g_sa1_1 = prep(sa1_1, 64, 64, 64);
  GL g_sa1_2 = prep(sa1_2, 64, 64, 128);
  GL g_sa2_0 = prep(sa2_0, 131, 160, 128);
  GL g_sa2_1 = prep(sa2_1, 128, 128, 128);
  GL g_sa2_2 = prep(sa2_2, 128, 128, 256);
  GL g_sa3_0 = prep(sa3_0, 259, 288, 256);
  GL g_sa3_1 = prep(sa3_1, 256, 256, 256);
  GL g_sa3_2 = prep(sa3_2, 256, 256, 512);
  GL g_sa4_0 = prep(sa4_0, 515, 544, 512);
  GL g_sa4_1 = prep(sa4_1, 512, 512, 512);
  GL g_sa4_2 = prep(sa4_2, 512, 512, 1024);
  GL g_fp4_0 = prep(fp4_0, 1536, 1536, 512);
  GL g_fp4_1 = prep(fp4_1, 512, 512, 512);
  GL g_fp3_0 = prep(fp3_0, 768, 768, 256);
  GL g_fp3_1 = prep(fp3_1, 256, 256, 256);
  GL g_fp2_0 = prep(fp2_0, 384, 384, 128);
  GL g_fp2_1 = prep(fp2_1, 128, 128, 128);
  GL g_fp1_0 = prep(fp1_0, 136, 160, 128);
  GL g_fp1_1 = prep(fp1_1, 128, 128, 128);
  GL g_head  = prep(head_l1, 128, 128, 128);

  auto gemm = [&](const float* X, const GL& g, float* Y, int R, int Kp, int O) {
    int waves = (R >> 4) * (O >> 6);
    gemm_bn_relu_wmma<<<(waves + 7) / 8, 256, 0, stream>>>(X, g.Wh, g.s, g.t, Y, R, Kp, O, 1);
  };

  split_kernel EW(Bn * N) (features, xyz, feats, Bn * N, CIN);

  // ---------------- SA1: 8192 -> 2048 pts, 32 nbrs, MLP 11->64->64->128 ----------------
  fps_kernel<<<Bn, 256, 0, stream>>>(xyz, N, 2048, fpsidx);
  gather_kernel EW(Bn * 2048 * 3) (xyz, fpsidx, l1_xyz, N, 2048, 3, Bn * 2048 * 3);
  knn_kernel<<<Bn * 2048, 256, (size_t)N * 4 + 256 * 8, stream>>>(l1_xyz, xyz, N, 2048, 32, knnidx);
  group_kernel EW(Bn * 2048 * 32) (xyz, feats, l1_xyz, knnidx, bufA, N, 2048, 32, INCH, 32, Bn * 2048 * 32);
  gemm(bufA, g_sa1_0, bufB, 262144, 32, 64);
  gemm(bufB, g_sa1_1, bufA, 262144, 64, 64);
  gemm(bufA, g_sa1_2, bufB, 262144, 64, 128);
  maxpool_kernel EW(Bn * 2048 * 128) (bufB, l1_pts, 2048, 32, 128, Bn * 2048 * 128);

  // ---------------- SA2: 2048 -> 512 pts, 64 nbrs, MLP 131->128->128->256 ----------------
  fps_kernel<<<Bn, 256, 0, stream>>>(l1_xyz, 2048, 512, fpsidx);
  gather_kernel EW(Bn * 512 * 3) (l1_xyz, fpsidx, l2_xyz, 2048, 512, 3, Bn * 512 * 3);
  knn_kernel<<<Bn * 512, 256, (size_t)2048 * 4 + 256 * 8, stream>>>(l2_xyz, l1_xyz, 2048, 512, 64, knnidx);
  group_kernel EW(Bn * 512 * 64) (l1_xyz, l1_pts, l2_xyz, knnidx, bufA, 2048, 512, 64, 128, 160, Bn * 512 * 64);
  gemm(bufA, g_sa2_0, bufB, 131072, 160, 128);
  gemm(bufB, g_sa2_1, bufA, 131072, 128, 128);
  gemm(bufA, g_sa2_2, bufB, 131072, 128, 256);
  maxpool_kernel EW(Bn * 512 * 256) (bufB, l2_pts, 512, 64, 256, Bn * 512 * 256);

  // ---------------- SA3: 512 -> 128 pts, 128 nbrs, MLP 259->256->256->512 ----------------
  fps_kernel<<<Bn, 256, 0, stream>>>(l2_xyz, 512, 128, fpsidx);
  gather_kernel EW(Bn * 128 * 3) (l2_xyz, fpsidx, l3_xyz, 512, 128, 3, Bn * 128 * 3);
  knn_kernel<<<Bn * 128, 256, (size_t)512 * 4 + 256 * 8, stream>>>(l3_xyz, l2_xyz, 512, 128, 128, knnidx);
  group_kernel EW(Bn * 128 * 128) (l2_xyz, l2_pts, l3_xyz, knnidx, bufA, 512, 128, 128, 256, 288, Bn * 128 * 128);
  gemm(bufA, g_sa3_0, bufB, 65536, 288, 256);
  gemm(bufB, g_sa3_1, bufA, 65536, 256, 256);
  gemm(bufA, g_sa3_2, bufB, 65536, 256, 512);
  maxpool_kernel EW(Bn * 128 * 512) (bufB, l3_pts, 128, 128, 512, Bn * 128 * 512);

  // ---------------- SA4: global group (B,1,128,515) padded to 544, MLP ->512->512->1024 ----------------
  copy_cols_kernel EW(Bn * 128 * 3)   (l3_xyz, bufA, 3, 544, 0, Bn * 128 * 3);
  copy_cols_kernel EW(Bn * 128 * 512) (l3_pts, bufA, 512, 544, 3, Bn * 128 * 512);
  zero_pad_kernel  EW(Bn * 128 * 29)  (bufA, 515, 544, Bn * 128 * 29);
  gemm(bufA, g_sa4_0, bufB, 512, 544, 512);
  gemm(bufB, g_sa4_1, bufA, 512, 512, 512);
  gemm(bufA, g_sa4_2, bufB, 512, 512, 1024);
  maxpool_kernel EW(Bn * 1024) (bufB, l4_pts, 1, 128, 1024, Bn * 1024);

  // ---------------- channel attention on l4 ----------------
  small_linear_kernel EW(Bn * 512) (l4_pts, attn_l1.W, attn_l1.b, attn_l1.gamma, attn_l1.beta,
                                    attn_l1.mean, attn_l1.var, attn_h, Bn, 1024, 512, 1);
  small_linear_kernel EW(Bn * 1024) (attn_h, attn2_W, attn2_b, nullptr, nullptr, nullptr, nullptr,
                                     attn_a, Bn, 512, 1024, 2);
  mul_kernel EW(Bn * 1024) (l4_pts, attn_a, Bn * 1024);

  // ---------------- FP4: broadcast l4 -> 128 pts, MLP 1536->512->512 ----------------
  copy_cols_kernel  EW(Bn * 128 * 512)  (l3_pts, bufA, 512, 1536, 0, Bn * 128 * 512);
  bcast_cols_kernel EW(Bn * 128 * 1024) (l4_pts, bufA, 1024, 1536, 512, 128, Bn * 128 * 1024);
  gemm(bufA, g_fp4_0, bufB, 512, 1536, 512);
  gemm(bufB, g_fp4_1, l3f, 512, 512, 512);

  // ---------------- FP3: 128 -> 512 pts, MLP 768->256->256 ----------------
  copy_cols_kernel EW(Bn * 512 * 256) (l2_pts, bufA, 256, 768, 0, Bn * 512 * 256);
  interp_kernel EW(Bn * 512) (l2_xyz, l3_xyz, l3f, bufA, 512, 128, 512, 768, 256, Bn * 512);
  gemm(bufA, g_fp3_0, bufB, 2048, 768, 256);
  gemm(bufB, g_fp3_1, l2f, 2048, 256, 256);

  // ---------------- FP2: 512 -> 2048 pts, MLP 384->128->128 ----------------
  copy_cols_kernel EW(Bn * 2048 * 128) (l1_pts, bufA, 128, 384, 0, Bn * 2048 * 128);
  interp_kernel EW(Bn * 2048) (l1_xyz, l2_xyz, l2f, bufA, 2048, 512, 256, 384, 128, Bn * 2048);
  gemm(bufA, g_fp2_0, bufB, 8192, 384, 128);
  gemm(bufB, g_fp2_1, l1f, 8192, 128, 128);

  // ---------------- FP1: 2048 -> 8192 pts, MLP 136 (pad 160) ->128->128 ----------------
  copy_cols_kernel EW(Bn * N * INCH) (feats, bufA, INCH, 160, 0, Bn * N * INCH);
  interp_kernel EW(Bn * N) (xyz, l1_xyz, l1f, bufA, N, 2048, 128, 160, INCH, Bn * N);
  zero_pad_kernel EW(Bn * N * 24) (bufA, 136, 160, Bn * N * 24);
  gemm(bufA, g_fp1_0, bufB, 32768, 160, 128);
  gemm(bufB, g_fp1_1, bufA, 32768, 128, 128);

  // ---------------- head: BN-ReLU 128->128 then linear 128->2 ----------------
  gemm(bufA, g_head, bufB, 32768, 128, 128);
  small_linear_kernel EW(32768 * 2) (bufB, head2_W, head2_b, nullptr, nullptr, nullptr, nullptr,
                                     (float*)d_out, 32768, 128, 2, 0);
}
